// IPAFluxAttnProcessor2_0_61993557950495
// MI455X (gfx1250) — matrix operand order repaired and checked by reading, not compiled, not run on hardware
//
#include <hip/hip_runtime.h>
#include <hip/hip_bf16.h>
#include <math.h>

// ---------------------------------------------------------------------------
// IPAFlux attention for MI455X (gfx1250), wave32 + WMMA bf16 (f32 accumulate).
//
// Roofline: ~270 GFLOP of GEMM/attention vs ~340MB of weight traffic
// (15us @ 23.3 TB/s) -> compute bound -> route all matmuls through
// v_wmma_f32_16x16x32_bf16 with fp32->bf16 conversion in-register while
// packing the documented per-lane fragment layouts (ISA 7.12.2).
//
// GEMM: 64x64 tile per wave (16 WMMAs per 32-deep k-step, 4 loads/WMMA),
// branchless row clamping for ragged M (no EXEC-predicated loads in the hot
// loop), global_prefetch on the operand streams.
// Attention: flash-style, V stored d-major so P@V B-fragments load as b64.
// Workspace needed: ~221 MB fp32.
// ---------------------------------------------------------------------------

#define H_      24
#define D_      128
#define HID_    3072
#define S_IMG_  2048
#define S_TXT_  512
#define S_TOT_  2560
#define EPS_    1e-5f
#define SM_SCALE 0.08838834764831845f   // 1/sqrt(128)

typedef __attribute__((ext_vector_type(16))) __bf16 v16bf;
typedef __attribute__((ext_vector_type(8)))  float  v8f;

__device__ __forceinline__ v8f wmma_bf16(v16bf a, v16bf b, v8f c) {
  // D = A(16x32 bf16) x B(32x16 bf16) + C(16x16 f32)
  return __builtin_amdgcn_wmma_f32_16x16x32_bf16(false, a, false, b,
                                                 (short)0, c, false, false);
}

// A-fragment K index for pair p (ISA 7.12.2, 16-bit A 16x32)
__device__ __forceinline__ int a_kidx(int p, int half) {
  return (p < 4) ? (half * 8 + 2 * p) : (16 + half * 8 + 2 * (p - 4));
}
// B-fragment K index for pair p (16-bit B 32x16, same K order as A)
__device__ __forceinline__ int b_kidx(int p, int half) {
  return (p < 4) ? (half * 8 + 2 * p) : (16 + half * 8 + 2 * (p - 4));
}

__device__ __forceinline__ float red_sum16(float v) {
#pragma unroll
  for (int m = 1; m <= 8; m <<= 1) v += __shfl_xor(v, m);
  return v;
}
__device__ __forceinline__ float red_max16(float v) {
#pragma unroll
  for (int m = 1; m <= 8; m <<= 1) v = fmaxf(v, __shfl_xor(v, m));
  return v;
}
__device__ __forceinline__ float red_sum32(float v) {
#pragma unroll
  for (int m = 1; m <= 16; m <<= 1) v += __shfl_xor(v, m);
  return v;
}

// ---------------------------------------------------------------------------
// out[M,N] = A[M,K] @ W[N,K]^T + bias[N]   (fp32 in/out, bf16 WMMA inside)
// block = 128 threads (4 waves) -> 128x128 block tile, 64x64 per wave.
// Ragged M handled by clamping the A row index (garbage rows never stored).
// ---------------------------------------------------------------------------
__global__ __launch_bounds__(128) void gemm_bf16_wmma(
    const float* __restrict__ A, const float* __restrict__ W,
    const float* __restrict__ bias, float* __restrict__ out,
    int M, int N, int K) {
  const int lane = threadIdx.x & 31;
  const int wave = threadIdx.x >> 5;
  const int half = lane >> 4;
  const int l16  = lane & 15;
  const int m0 = blockIdx.y * 128 + (wave >> 1) * 64;
  const int n0 = blockIdx.x * 128 + (wave & 1) * 64;

  const float* Arow[4];
  const float* Wrow[4];
#pragma unroll
  for (int i = 0; i < 4; ++i) {
    int m = m0 + i * 16 + l16;
    m = (m < M) ? m : (M - 1);            // branchless clamp: loads always valid
    Arow[i] = A + (size_t)m * K;
    Wrow[i] = W + (size_t)(n0 + i * 16 + l16) * K;   // N % 128 == 0
  }

  v8f acc[4][4];
  const v8f vz = {};
#pragma unroll
  for (int i = 0; i < 4; ++i)
#pragma unroll
    for (int j = 0; j < 4; ++j) acc[i][j] = vz;

  for (int k = 0; k < K; k += 32) {
    v16bf af[4], bf[4];
#pragma unroll
    for (int i = 0; i < 4; ++i) {
#pragma unroll
      for (int p = 0; p < 8; ++p) {
        const int ka = a_kidx(p, half);
        af[i][2 * p]     = (__bf16)Arow[i][k + ka];
        af[i][2 * p + 1] = (__bf16)Arow[i][k + ka + 1];
        const int kb = b_kidx(p, half);
        bf[i][2 * p]     = (__bf16)Wrow[i][k + kb];
        bf[i][2 * p + 1] = (__bf16)Wrow[i][k + kb + 1];
      }
      // keep the operand streams ahead of the math (global_prefetch_b8)
      __builtin_prefetch(Arow[i] + k + 256, 0, 1);
      __builtin_prefetch(Wrow[i] + k + 256, 0, 1);
    }
#pragma unroll
    for (int i = 0; i < 4; ++i)
#pragma unroll
      for (int j = 0; j < 4; ++j)
        acc[i][j] = wmma_bf16(af[i], bf[j], acc[i][j]);
  }

  float bn[4];
#pragma unroll
  for (int j = 0; j < 4; ++j) bn[j] = bias ? bias[n0 + j * 16 + l16] : 0.f;

#pragma unroll
  for (int i = 0; i < 4; ++i) {
#pragma unroll
    for (int r = 0; r < 8; ++r) {
      const int mr = m0 + i * 16 + half * 8 + r;   // C/D: VGPR r -> row r+8*half
      if (mr < M) {
#pragma unroll
        for (int j = 0; j < 4; ++j)
          out[(size_t)mr * N + n0 + j * 16 + l16] = acc[i][j][r] + bn[j];
      }
    }
  }
}

// ---------------------------------------------------------------------------
// Build head-major Q/K [H, S_TOT, D] and d-major V [H, D, S_TOT] (all fp32):
// rows < S_TXT from encoder projections, rest from image projections.
// q,k: per-head RMSNorm then RoPE; v: transpose copy. One wave per (s,h).
// Each lane owns RoPE pairs (2*lane, 2*lane+1) and (2*lane+64, 2*lane+65).
// ---------------------------------------------------------------------------
__global__ __launch_bounds__(256) void build_qkv(
    const float* __restrict__ qi, const float* __restrict__ ki,
    const float* __restrict__ vi, const float* __restrict__ eq,
    const float* __restrict__ ek, const float* __restrict__ ev,
    const float* __restrict__ cosr, const float* __restrict__ sinr,
    float* __restrict__ Qf, float* __restrict__ Kf, float* __restrict__ Vt) {
  const int lane = threadIdx.x & 31;
  const int g = blockIdx.x * (blockDim.x >> 5) + (threadIdx.x >> 5);
  if (g >= S_TOT_ * H_) return;
  const int s = g / H_;
  const int h = g % H_;

  const float *qs, *ks, *vs;
  int row;
  if (s < S_TXT_) { qs = eq; ks = ek; vs = ev; row = s; }
  else            { qs = qi; ks = ki; vs = vi; row = s - S_TXT_; }

  const int d0 = 2 * lane;                       // d0, d0+1, d0+64, d0+65
  const size_t ib = (size_t)row * HID_ + h * D_;

  float q0 = qs[ib + d0], q1 = qs[ib + d0 + 1];
  float q2 = qs[ib + d0 + 64], q3 = qs[ib + d0 + 65];
  float k0 = ks[ib + d0], k1 = ks[ib + d0 + 1];
  float k2 = ks[ib + d0 + 64], k3 = ks[ib + d0 + 65];
  float v0 = vs[ib + d0], v1 = vs[ib + d0 + 1];
  float v2 = vs[ib + d0 + 64], v3 = vs[ib + d0 + 65];

  const float sq = red_sum32(q0 * q0 + q1 * q1 + q2 * q2 + q3 * q3);
  const float sk = red_sum32(k0 * k0 + k1 * k1 + k2 * k2 + k3 * k3);
  const float nq = rsqrtf(sq * (1.f / 128.f) + EPS_);
  const float nk = rsqrtf(sk * (1.f / 128.f) + EPS_);
  q0 *= nq; q1 *= nq; q2 *= nq; q3 *= nq;
  k0 *= nk; k1 *= nk; k2 *= nk; k3 *= nk;

  const size_t rb = (size_t)s * D_;
  const float c0 = cosr[rb + d0], c1 = cosr[rb + d0 + 1];
  const float c2 = cosr[rb + d0 + 64], c3 = cosr[rb + d0 + 65];
  const float t0 = sinr[rb + d0], t1 = sinr[rb + d0 + 1];
  const float t2 = sinr[rb + d0 + 64], t3 = sinr[rb + d0 + 65];

  // rope: even: x*cos - x_odd*sin ; odd: x*cos + x_even*sin
  const float qo0 = q0 * c0 - q1 * t0, qo1 = q1 * c1 + q0 * t1;
  const float qo2 = q2 * c2 - q3 * t2, qo3 = q3 * c3 + q2 * t3;
  const float ko0 = k0 * c0 - k1 * t0, ko1 = k1 * c1 + k0 * t1;
  const float ko2 = k2 * c2 - k3 * t2, ko3 = k3 * c3 + k2 * t3;

  const size_t ob = ((size_t)h * S_TOT_ + s) * D_;
  Qf[ob + d0] = qo0; Qf[ob + d0 + 1] = qo1;
  Qf[ob + d0 + 64] = qo2; Qf[ob + d0 + 65] = qo3;
  Kf[ob + d0] = ko0; Kf[ob + d0 + 1] = ko1;
  Kf[ob + d0 + 64] = ko2; Kf[ob + d0 + 65] = ko3;

  // V transposed: Vt[h][d][s]
  const size_t vb = ((size_t)h * D_) * S_TOT_ + s;
  Vt[vb + (size_t)(d0)      * S_TOT_] = v0;
  Vt[vb + (size_t)(d0 + 1)  * S_TOT_] = v1;
  Vt[vb + (size_t)(d0 + 64) * S_TOT_] = v2;
  Vt[vb + (size_t)(d0 + 65) * S_TOT_] = v3;
}

// ---------------------------------------------------------------------------
// IP k/v head prep: kipn[h,r,:] = rmsnorm(kraw[r, h*128:]) ; vip copy.
// One wave per (h,r). 96 waves total.
// ---------------------------------------------------------------------------
__global__ __launch_bounds__(256) void build_ip_heads(
    const float* __restrict__ kraw, const float* __restrict__ vraw,
    float* __restrict__ kipn, float* __restrict__ vip) {
  const int lane = threadIdx.x & 31;
  const int g = blockIdx.x * (blockDim.x >> 5) + (threadIdx.x >> 5);
  if (g >= H_ * 4) return;
  const int h = g >> 2, r = g & 3;
  const int d0 = 2 * lane;
  const size_t ib = (size_t)r * HID_ + h * D_;

  float k0 = kraw[ib + d0], k1 = kraw[ib + d0 + 1];
  float k2 = kraw[ib + d0 + 64], k3 = kraw[ib + d0 + 65];
  const float ss = red_sum32(k0 * k0 + k1 * k1 + k2 * k2 + k3 * k3);
  const float nk = rsqrtf(ss * (1.f / 128.f) + EPS_);

  const size_t ob = (size_t)g * D_;
  kipn[ob + d0] = k0 * nk; kipn[ob + d0 + 1] = k1 * nk;
  kipn[ob + d0 + 64] = k2 * nk; kipn[ob + d0 + 65] = k3 * nk;
  vip[ob + d0] = vraw[ib + d0]; vip[ob + d0 + 1] = vraw[ib + d0 + 1];
  vip[ob + d0 + 64] = vraw[ib + d0 + 64]; vip[ob + d0 + 65] = vraw[ib + d0 + 65];
}

// ---------------------------------------------------------------------------
// IP attention: 4 keys per head -> plain VALU per (h, m) query row.
// Accumulates IP_SCALE * ip_out into attn rows [S_TXT .. S_TOT).
// q is re-normalized here (pre-RoPE q, per reference).
// ---------------------------------------------------------------------------
__global__ __launch_bounds__(256) void ip_attn(
    const float* __restrict__ qi, const float* __restrict__ kipn,
    const float* __restrict__ vip, float* __restrict__ attn) {
  const int t = blockIdx.x * blockDim.x + threadIdx.x;
  if (t >= H_ * S_IMG_) return;
  const int h = t / S_IMG_, m = t % S_IMG_;
  const float* q = qi + (size_t)m * HID_ + h * D_;

  float ss = 0.f;
  for (int d = 0; d < D_; ++d) { const float x = q[d]; ss += x * x; }
  const float scale = rsqrtf(ss * (1.f / 128.f) + EPS_) * SM_SCALE;

  float sc[4];
#pragma unroll
  for (int j = 0; j < 4; ++j) {
    const float* kk = kipn + ((size_t)(h * 4 + j)) * D_;
    float dot = 0.f;
    for (int d = 0; d < D_; ++d) dot += q[d] * kk[d];
    sc[j] = dot * scale;
  }
  float mx = fmaxf(fmaxf(sc[0], sc[1]), fmaxf(sc[2], sc[3]));
  float p[4], sum = 0.f;
#pragma unroll
  for (int j = 0; j < 4; ++j) { p[j] = __expf(sc[j] - mx); sum += p[j]; }
  const float inv = 1.f / sum;   // IP_SCALE == 1.0

  float* o = attn + ((size_t)(S_TXT_ + m)) * HID_ + h * D_;
  const float* vb = vip + (size_t)(h * 4) * D_;
  for (int d = 0; d < D_; ++d) {
    const float a = p[0] * vb[d] + p[1] * vb[D_ + d] +
                    p[2] * vb[2 * D_ + d] + p[3] * vb[3 * D_ + d];
    o[d] += a * inv;
  }
}

// ---------------------------------------------------------------------------
// Flash attention: Q/K [H, S_TOT, D], V d-major [H, D, S_TOT] (fp32 in,
// bf16 WMMA inside). block = 128 threads (4 waves); wave owns a 16-row Q tile.
// Per 32-key step: 8 WMMAs for S = QK^T, online softmax (row reductions over
// the 16-lane half-groups of the C layout), P repacked via LDS into A layout,
// 8 WMMAs for O += P@V. Softmax scale folded into the Q->bf16 conversion.
// ---------------------------------------------------------------------------
__global__ __launch_bounds__(128) void flash_attn(
    const float* __restrict__ Qf, const float* __restrict__ Kf,
    const float* __restrict__ Vt, float* __restrict__ attn) {
  __shared__ float pl[4 * 16 * 32];          // per-wave 16x32 P tile
  const int lane = threadIdx.x & 31;
  const int wave = threadIdx.x >> 5;
  const int half = lane >> 4;
  const int l16  = lane & 15;
  const int h  = blockIdx.y;
  const int q0 = (blockIdx.x * 4 + wave) * 16;
  const size_t hb = (size_t)h * S_TOT_;
  float* myp = pl + wave * 512;

  // Q fragments for the 4 K-chunks of D=128 (scale folded in)
  v16bf qa[4];
  const float* Qrow = Qf + (hb + q0 + l16) * (size_t)D_;
#pragma unroll
  for (int c = 0; c < 4; ++c) {
#pragma unroll
    for (int p = 0; p < 8; ++p) {
      const int ka = a_kidx(p, half);
      qa[c][2 * p]     = (__bf16)(Qrow[c * 32 + ka]     * SM_SCALE);
      qa[c][2 * p + 1] = (__bf16)(Qrow[c * 32 + ka + 1] * SM_SCALE);
    }
  }

  v8f oc[8];
  const v8f vz = {};
#pragma unroll
  for (int c = 0; c < 8; ++c) oc[c] = vz;
  float Mst[8], Lst[8];
#pragma unroll
  for (int r = 0; r < 8; ++r) { Mst[r] = -1e30f; Lst[r] = 0.f; }

  for (int j0 = 0; j0 < S_TOT_; j0 += 32) {
    // ---- S = Q @ K^T (two 16-key subtiles) ----
    v8f s0 = vz, s1 = vz;
#pragma unroll
    for (int c = 0; c < 4; ++c) {
      v16bf b0, b1;
      const float* K0 = Kf + (hb + j0 + l16) * (size_t)D_ + c * 32;
      const float* K1 = K0 + 16 * D_;
#pragma unroll
      for (int p = 0; p < 8; ++p) {
        const int kb = b_kidx(p, half);
        b0[2 * p]     = (__bf16)K0[kb];
        b0[2 * p + 1] = (__bf16)K0[kb + 1];
        b1[2 * p]     = (__bf16)K1[kb];
        b1[2 * p + 1] = (__bf16)K1[kb + 1];
      }
      s0 = wmma_bf16(qa[c], b0, s0);
      s1 = wmma_bf16(qa[c], b1, s1);
    }

    // ---- online softmax (row m = r + 8*half lives in a 16-lane half-group) --
#pragma unroll
    for (int r = 0; r < 8; ++r) {
      float mx = red_max16(fmaxf(s0[r], s1[r]));
      const float newM = fmaxf(Mst[r], mx);
      const float corr = __expf(Mst[r] - newM);
      const float p0 = __expf(s0[r] - newM);
      const float p1 = __expf(s1[r] - newM);
      const float rs = red_sum16(p0 + p1);
      Lst[r] = Lst[r] * corr + rs;
      Mst[r] = newM;
#pragma unroll
      for (int c = 0; c < 8; ++c) oc[c][r] *= corr;
      const int mrow = half * 8 + r;
      myp[mrow * 32 + l16]      = p0;
      myp[mrow * 32 + 16 + l16] = p1;
    }
    __syncthreads();

    // ---- repack P (C layout -> A layout) through LDS ----
    v16bf pa;
#pragma unroll
    for (int p = 0; p < 8; ++p) {
      const int ka = a_kidx(p, half);
      pa[2 * p]     = (__bf16)myp[l16 * 32 + ka];
      pa[2 * p + 1] = (__bf16)myp[l16 * 32 + ka + 1];
    }
    __syncthreads();

    // ---- O += P @ V  (V d-major: pairs along keys are contiguous -> b64) ----
#pragma unroll
    for (int c = 0; c < 8; ++c) {
      v16bf bv;
      const float* Vrow = Vt + ((size_t)h * D_ + c * 16 + l16) * S_TOT_ + j0;
#pragma unroll
      for (int p = 0; p < 8; ++p) {
        const int kb = b_kidx(p, half);
        bv[2 * p]     = (__bf16)Vrow[kb];
        bv[2 * p + 1] = (__bf16)Vrow[kb + 1];
      }
      oc[c] = wmma_bf16(pa, bv, oc[c]);
    }
  }

  // ---- epilogue: normalize and write un-headed [S_TOT, HID] fp32 ----
#pragma unroll
  for (int r = 0; r < 8; ++r) {
    const float inv = 1.f / Lst[r];
    const int srow = q0 + half * 8 + r;
#pragma unroll
    for (int c = 0; c < 8; ++c) {
      attn[(size_t)srow * HID_ + h * D_ + c * 16 + l16] = oc[c][r] * inv;
    }
  }
}

// ---------------------------------------------------------------------------
extern "C" void kernel_launch(void* const* d_in, const int* in_sizes, int n_in,
                              void* d_out, int out_size, void* d_ws,
                              size_t ws_size, hipStream_t stream) {
  (void)in_sizes; (void)n_in; (void)out_size; (void)ws_size;

  const float* hidden = (const float*)d_in[0];
  const float* enc    = (const float*)d_in[1];
  const float* img    = (const float*)d_in[2];
  const float* cosr   = (const float*)d_in[3];
  const float* sinr   = (const float*)d_in[4];
  const float* Wq  = (const float*)d_in[5];  const float* bq  = (const float*)d_in[6];
  const float* Wk  = (const float*)d_in[7];  const float* bk  = (const float*)d_in[8];
  const float* Wv  = (const float*)d_in[9];  const float* bv  = (const float*)d_in[10];
  const float* Waq = (const float*)d_in[11]; const float* baq = (const float*)d_in[12];
  const float* Wak = (const float*)d_in[13]; const float* bak = (const float*)d_in[14];
  const float* Wav = (const float*)d_in[15]; const float* bav = (const float*)d_in[16];
  const float* Wkip = (const float*)d_in[17];
  const float* Wvip = (const float*)d_in[18];
  const float* Wo  = (const float*)d_in[19]; const float* bo  = (const float*)d_in[20];
  const float* Wao = (const float*)d_in[21]; const float* bao = (const float*)d_in[22];

  float* ws = (float*)d_ws;
  const size_t oi = (size_t)S_IMG_ * HID_;     // 6291456
  const size_t oe = (size_t)S_TXT_ * HID_;     // 1572864
  const size_t oq = (size_t)H_ * S_TOT_ * D_;  // 7864320
  float* q_img = ws;
  float* k_img = q_img + oi;
  float* v_img = k_img + oi;
  float* eq    = v_img + oi;
  float* ek    = eq + oe;
  float* ev    = ek + oe;
  float* kraw  = ev + oe;            // 4*HID
  float* vraw  = kraw + 4 * HID_;
  float* kipn  = vraw + 4 * HID_;    // 24*4*128 == 4*HID
  float* vip   = kipn + 4 * HID_;
  float* Qf    = vip + 4 * HID_;
  float* Kf    = Qf + oq;
  float* Vt    = Kf + oq;            // [H, D, S_TOT]
  float* attn  = Vt + oq;            // S_TOT * HID

  const dim3 blk128(128);
  auto gemm = [&](const float* A, const float* W, const float* b, float* o,
                  int M, int N) {
    dim3 grid(N / 128, (M + 127) / 128);
    gemm_bf16_wmma<<<grid, blk128, 0, stream>>>(A, W, b, o, M, N, HID_);
  };

  // Projections
  gemm(hidden, Wq, bq, q_img, S_IMG_, HID_);
  gemm(hidden, Wk, bk, k_img, S_IMG_, HID_);
  gemm(hidden, Wv, bv, v_img, S_IMG_, HID_);
  gemm(enc, Waq, baq, eq, S_TXT_, HID_);
  gemm(enc, Wak, bak, ek, S_TXT_, HID_);
  gemm(enc, Wav, bav, ev, S_TXT_, HID_);
  gemm(img, Wkip, nullptr, kraw, 4, HID_);
  gemm(img, Wvip, nullptr, vraw, 4, HID_);

  // Head prep
  build_ip_heads<<<12, 256, 0, stream>>>(kraw, vraw, kipn, vip);
  build_qkv<<<(S_TOT_ * H_) / 8, 256, 0, stream>>>(
      q_img, k_img, v_img, eq, ek, ev, cosr, sinr, Qf, Kf, Vt);

  // Attention (writes attn fully) then IP attention (accumulates into it)
  flash_attn<<<dim3(40, H_), blk128, 0, stream>>>(Qf, Kf, Vt, attn);
  ip_attn<<<(H_ * S_IMG_ + 255) / 256, 256, 0, stream>>>(q_img, kipn, vip, attn);

  // Output projections straight into d_out: hid_out then enc_out
  float* outp = (float*)d_out;
  gemm(attn + (size_t)S_TXT_ * HID_, Wo, bo, outp, S_IMG_, HID_);
  gemm(attn, Wao, bao, outp + (size_t)S_IMG_ * HID_, S_TXT_, HID_);
}
